// BayesConv2dMF_11639361372311
// MI455X (gfx1250) — matrix operand order repaired and checked by reading, not compiled, not run on hardware
//
#include <hip/hip_runtime.h>
#include <hip/hip_bf16.h>

typedef __attribute__((ext_vector_type(16))) __bf16 v16bf;
typedef __attribute__((ext_vector_type(8)))  float  v8f;

#define B_    32
#define CO_   128
#define CI_   128
#define H_    56
#define W_    56
#define HW_   (H_ * W_)          // 3136
#define NTILE 64                 // pixel tile per block (3136 / 64 = 49 exact)
#define KC    32                 // ci chunk per WMMA K-step
#define HROWS 5                  // halo rows: up to 3 image rows + 2 halo
#define HCOLS 58                 // 56 cols + left/right halo
#define HSLICE (HCOLS * KC)      // elements per halo row
#define HUNITS (HROWS * HCOLS * 4)  // fill units of 8 ci each = 1160
#define BLK   128                // 4 waves: wave grid 4(M) x 1(N)

// 32-byte fragment assembly: element order matches memory order
union frag16 { uint4 u4[2]; v16bf v; };
union pack8  { uint4 u4; __bf16 e[8]; };

// ---------------------------------------------------------------------------
// Stage 1: w = eps * exp(psi) + mu  ->  bf16, layout [b][rs][co][ci]
// so each (b, rs) slice is a row-major M=co x K=ci A-matrix for WMMA.
// ---------------------------------------------------------------------------
__global__ __launch_bounds__(256) void weight_prep_kernel(
    const float* __restrict__ eps,
    const float* __restrict__ psi,
    const float* __restrict__ mu,
    __bf16* __restrict__ wbf)
{
    int t  = blockIdx.x * 256 + threadIdx.x;        // == output index
    int ci = t & (CI_ - 1);
    int co = (t >> 7) & (CO_ - 1);
    int rs = (t >> 14) % 9;
    int b  = t / (9 << 14);

    int pidx = (co * CI_ + ci) * 9 + rs;                    // [co][ci][kh][kw]
    int eidx = ((b * CO_ + co) * CI_ + ci) * 9 + rs;        // [b][co][ci][kh][kw]

    float w = eps[eidx] * __expf(psi[pidx]) + mu[pidx];
    wbf[t] = (__bf16)w;
}

// ---------------------------------------------------------------------------
// Stage 2: implicit-GEMM conv as 9 shift-GEMMs of K=128, bf16 WMMA, f32 acc.
// kc-outer / rs-inner: one LDS halo tile ([5][58][32ci] bf16) per ci-chunk
// serves all 9 taps -> 72 WMMAs per wave per fill+barrier pair.
// Block: 128 thr (4 waves), tile M=128 (all co) x N=64 px, one batch sample.
// Wave grid 4(M) x 1(N); wave tile 32(M) x 64(N): 2 A-frags x 4 B-frags,
// 8 accumulators -> each A-frag load feeds 4 WMMAs, no cross-wave A dup.
// ---------------------------------------------------------------------------
__global__ __launch_bounds__(BLK) void conv_wmma_kernel(
    const float*  __restrict__ in,    // [B][CI][H][W]
    const __bf16* __restrict__ wbf,   // [B][9][CO][CI]
    float*        __restrict__ out)   // [B][CO][H][W]
{
    __shared__ __bf16 sH[HROWS * HSLICE];      // halo tile, 18.125 KB

    const int b    = blockIdx.y;
    const int n0   = blockIdx.x * NTILE;       // pixel tile base
    const int tid  = threadIdx.x;
    const int lane = tid & 31;
    const int waveM = tid >> 5;                // 0..3 -> co base = waveM*32
    const int lrow  = lane & 15;
    const int lhalf = lane >> 4;

    const int rfirst = n0 / W_;                // first image row of this tile

    // Per-lane LDS base offsets for the 4 N-fragments (B columns).
    // Halo coords: row (h - rfirst + r), col (w + s); +lhalf*16 picks K-half.
    int bbase[4];
    #pragma unroll
    for (int n = 0; n < 4; ++n) {
        int p = n0 + n * 16 + lrow;
        int h = p / W_;
        int w = p - h * W_;
        bbase[n] = ((h - rfirst) * HCOLS + w) * KC + lhalf * 16;
    }

    // Per-lane A row offsets (elements) for the 2 M-fragments.
    int arow[2];
    #pragma unroll
    for (int m = 0; m < 2; ++m)
        arow[m] = (waveM * 32 + m * 16 + lrow) * CI_ + lhalf * 16;

    v8f acc[2][4] = {};                        // 64 VGPRs of f32 accum

    const float*  inB = in  + (size_t)b * CI_ * HW_;
    const __bf16* wB  = wbf + (size_t)b * 9 * CO_ * CI_;

    for (int kc = 0; kc < CI_; kc += KC) {
        __syncthreads();   // previous chunk done consuming sH

        // ---- stage halo input tile (zero-padded) into LDS as bf16 ----
        for (int u = tid; u < HUNITS; u += BLK) {
            const int ci8 = (u & 3) * 8;
            const int hc  = (u >> 2) % HCOLS;
            const int hr  = (u >> 2) / HCOLS;
            const int gh  = rfirst - 1 + hr;
            const int gw  = hc - 1;
            const bool v  = (gh >= 0) & (gh < H_) & (gw >= 0) & (gw < W_);
            const float* src = inB + (size_t)(kc + ci8) * HW_ + gh * W_ + gw;
            pack8 pk;
            #pragma unroll
            for (int i = 0; i < 8; ++i)
                pk.e[i] = (__bf16)(v ? src[(size_t)i * HW_] : 0.0f);
            *(uint4*)(&sH[hr * HSLICE + hc * KC + ci8]) = pk.u4;

            // speculative prefetch of next ci-chunk (global_prefetch_b8);
            // dropped silently if address is bad
            if (kc + KC < CI_)
                __builtin_prefetch(src + (size_t)KC * HW_, 0, 1);
        }
        __syncthreads();

        // ---- 9 taps from the same halo tile: 72 WMMAs per wave per fill ----
        #pragma unroll
        for (int rs = 0; rs < 9; ++rs) {
            const int r = rs / 3;
            const int s = rs - 3 * r;
            const __bf16* wA = wB + (size_t)rs * CO_ * CI_ + kc;

            // A fragments (global bf16, row-major [co][ci], L2-resident)
            v16bf afrag[2];
            #pragma unroll
            for (int m = 0; m < 2; ++m) {
                const __bf16* ap = wA + arow[m];
                frag16 f;
                f.u4[0] = *(const uint4*)(ap);        // K = base + 0..7
                f.u4[1] = *(const uint4*)(ap + 8);    // K = base + 8..15
                afrag[m] = f.v;
            }

            // B fragments (LDS halo tile, 16 contiguous ci per lane)
            v16bf bfrag[4];
            #pragma unroll
            for (int n = 0; n < 4; ++n) {
                const __bf16* bp = &sH[bbase[n] + (r * HCOLS + s) * KC];
                frag16 f;
                f.u4[0] = *(const uint4*)(bp);        // K = lhalf*16 + 0..7
                f.u4[1] = *(const uint4*)(bp + 8);    // K = lhalf*16 + 8..15
                bfrag[n] = f.v;
            }

            // 8 WMMAs: 32x64 wave tile
            #pragma unroll
            for (int m = 0; m < 2; ++m)
                #pragma unroll
                for (int n = 0; n < 4; ++n)
                    acc[m][n] = __builtin_amdgcn_wmma_f32_16x16x32_bf16(
                        false, afrag[m], false, bfrag[n],
                        (short)0, acc[m][n], false, false);
        }
    }

    // ---- store C tiles: VGPR i -> M = i + 8*lhalf, N = lrow ----
    float* outB = out + (size_t)b * CO_ * HW_;
    #pragma unroll
    for (int m = 0; m < 2; ++m) {
        #pragma unroll
        for (int n = 0; n < 4; ++n) {
            const int px = n0 + n * 16 + lrow;
            #pragma unroll
            for (int i = 0; i < 8; ++i) {
                const int co = waveM * 32 + m * 16 + lhalf * 8 + i;
                outB[(size_t)co * HW_ + px] = acc[m][n][i];
            }
        }
    }
}

// ---------------------------------------------------------------------------
extern "C" void kernel_launch(void* const* d_in, const int* in_sizes, int n_in,
                              void* d_out, int out_size, void* d_ws, size_t ws_size,
                              hipStream_t stream) {
    const float* input = (const float*)d_in[0];   // [B][CI][H][W]
    const float* eps   = (const float*)d_in[1];   // [B][CO][CI][3][3]
    const float* psi   = (const float*)d_in[2];   // [CO][CI][3][3]
    const float* mu    = (const float*)d_in[3];   // [CO][CI][3][3]
    float*   out = (float*)d_out;                 // [B][CO][H][W]
    __bf16*  wbf = (__bf16*)d_ws;                 // [B][9][CO][CI] bf16 (9.4 MB)

    const int nw = B_ * 9 * CO_ * CI_;            // 4,718,592 (multiple of 256)
    weight_prep_kernel<<<dim3(nw / 256), dim3(256), 0, stream>>>(eps, psi, mu, wbf);

    conv_wmma_kernel<<<dim3(HW_ / NTILE, B_), dim3(BLK), 0, stream>>>(input, wbf, out);
}
